// Decoder_63316407878002
// MI455X (gfx1250) — matrix operand order repaired and checked by reading, not compiled, not run on hardware
//
#include <hip/hip_runtime.h>
#include <hip/hip_bf16.h>

typedef __attribute__((ext_vector_type(16))) __bf16 bf16x16;
typedef __attribute__((ext_vector_type(8)))  float  floatx8;

#define Bb 256
#define Tt 512
#define Dd 512
#define Nn 64

union BFrag { bf16x16 v; uint4 q[2]; };

// A-fragment (16x32, bf16) from row-major memory. ISA layout:
// lane L: row M = L%16; comps 0-7 = K(koff..koff+7), comps 8-15 = K(16+koff..), koff = (L<16)?0:8
__device__ __forceinline__ BFrag load_a_rowmajor(const __bf16* tile, int ld, int lane) {
    int m = lane & 15;
    int koff = (lane & 16) ? 8 : 0;
    const __bf16* p = tile + (size_t)m * ld + koff;
    BFrag f;
    f.q[0] = *(const uint4*)(p);
    f.q[1] = *(const uint4*)(p + 16);
    return f;
}

// B-fragment (32x16, bf16) from pre-transposed weight Wt[K][Nout] (n contiguous).
// ISA layout: lane L holds N = nb+L%16; comps 0-7 = K(k0+koff..+7), comps 8-15 = K(k0+16+koff..), koff = (L<16)?0:8
__device__ __forceinline__ BFrag load_b_kmajor(const __bf16* wt, int ldn, int k0, int nb, int lane) {
    int koff = (lane & 16) ? 8 : 0;
    const __bf16* p = wt + (size_t)(k0 + koff) * ldn + nb + (lane & 15);
    BFrag f;
#pragma unroll
    for (int s = 0; s < 8; ++s) {
        f.v[s] = p[(size_t)s * ldn];
        f.v[s + 8] = p[(size_t)(s + 16) * ldn];
    }
    return f;
}

__device__ __forceinline__ floatx8 wmma_bf16(BFrag a, BFrag b, floatx8 c) {
    return __builtin_amdgcn_wmma_f32_16x16x32_bf16(false, a.v, false, b.v, (short)0, c, false, false);
}

__device__ __forceinline__ float sigmoid_f(float x) {
    return 1.0f / (1.0f + __expf(-x));
}
__device__ __forceinline__ float tanh_f(float x) {
    x = fminf(fmaxf(x, -15.0f), 15.0f);
    float e2 = __expf(2.0f * x);
    return (e2 - 1.0f) / (e2 + 1.0f);
}

// Compute y = softmax(h[rb:rb+16] @ Wlin^T + b_lin) into slog (fp32) and sy (bf16).
__device__ __forceinline__ void compute_y_tile(
    const __bf16* __restrict__ hread, const __bf16* __restrict__ WlinT,
    const float* __restrict__ b_lin,
    int rb, int wave, int lane, int tid,
    float (*slog)[64], __bf16 (*sy)[64])
{
    floatx8 acc = {};
    int nb = wave * 16;
    const __bf16* atile = hread + (size_t)rb * Dd;
    for (int k0 = 0; k0 < Dd; k0 += 32) {
        BFrag a = load_a_rowmajor(atile + k0, Dd, lane);
        BFrag b = load_b_kmajor(WlinT, Nn, k0, nb, lane);
        acc = wmma_bf16(a, b, acc);
    }
    int n = nb + (lane & 15);
    float bl = b_lin[n];
    int mbase = (lane & 16) ? 8 : 0;
#pragma unroll
    for (int r = 0; r < 8; ++r) slog[r + mbase][n] = acc[r] + bl;
    __syncthreads();
    if (tid < 16) {
        float mx = -1e30f;
        for (int j = 0; j < 64; ++j) mx = fmaxf(mx, slog[tid][j]);
        float s = 0.0f;
        for (int j = 0; j < 64; ++j) s += __expf(slog[tid][j] - mx);
        float inv = 1.0f / s;
        for (int j = 0; j < 64; ++j) {
            float yv = __expf(slog[tid][j] - mx) * inv;
            slog[tid][j] = yv;
            sy[tid][j]   = (__bf16)yv;
        }
    }
    __syncthreads();
}

// One timestep. grid = (B/16, D/64), block = 128 (4 waves).
// Wave w handles cols [db + w*16, +16) of each gate i/f/g/o -> cell update fully in registers.
__global__ void __launch_bounds__(128)
lstm_step_kernel(const __bf16* __restrict__ xbf,    // [B][T][D] bf16
                 const __bf16* __restrict__ WihT,   // [D+N][4D] bf16 (transposed)
                 const __bf16* __restrict__ WhhT,   // [D][4D]  bf16 (transposed)
                 const __bf16* __restrict__ WlinT,  // [D][N]   bf16 (transposed)
                 const float*  __restrict__ bsum,   // [4D] = b_ih + b_hh
                 const float*  __restrict__ b_lin,  // [N]
                 const __bf16* __restrict__ hread,  // [B][D] bf16
                 __bf16*       __restrict__ hwrite,
                 const float*  __restrict__ cread,  // [B][D] fp32
                 float*        __restrict__ cwrite,
                 float*        __restrict__ out,    // [B][T][N] fp32
                 int t)
{
    __shared__ float  slog[16][64];
    __shared__ __align__(16) __bf16 sy[16][64];
    int tid  = threadIdx.x;
    int lane = tid & 31;
    int wave = tid >> 5;
    int rb = blockIdx.x * 16;
    int db = blockIdx.y * 64;

    // Stage 1: y(t-1) = softmax(h(t-1) @ Wlin^T + b_lin) for our 16 batch rows.
    if (t == 0) {
        for (int i = tid; i < 16 * 64; i += 128) sy[i >> 6][i & 63] = (__bf16)0.0f;
        __syncthreads();
    } else {
        compute_y_tile(hread, WlinT, b_lin, rb, wave, lane, tid, slog, sy);
        if (blockIdx.y == 0) {
            for (int i = tid; i < 16 * 64; i += 128) {
                int m = i >> 6, n = i & 63;
                out[((size_t)(rb + m) * Tt + (t - 1)) * Nn + n] = slog[m][n];
            }
        }
    }

    // Stage 2: gates = x_t@Wx^T + y@Wy^T + h@Whh^T (bias added after).
    floatx8 acc0 = {}, acc1 = {}, acc2 = {}, acc3 = {};
    int nb0 = 0 * Dd + db + wave * 16;
    int nb1 = 1 * Dd + db + wave * 16;
    int nb2 = 2 * Dd + db + wave * 16;
    int nb3 = 3 * Dd + db + wave * 16;

    // x part: K = 0..511 of W_ih (rows of WihT)
    const __bf16* xtile = xbf + ((size_t)rb * Tt + t) * Dd;
    for (int k0 = 0; k0 < Dd; k0 += 32) {
        BFrag a = load_a_rowmajor(xtile + k0, Tt * Dd, lane);
        acc0 = wmma_bf16(a, load_b_kmajor(WihT, 4 * Dd, k0, nb0, lane), acc0);
        acc1 = wmma_bf16(a, load_b_kmajor(WihT, 4 * Dd, k0, nb1, lane), acc1);
        acc2 = wmma_bf16(a, load_b_kmajor(WihT, 4 * Dd, k0, nb2, lane), acc2);
        acc3 = wmma_bf16(a, load_b_kmajor(WihT, 4 * Dd, k0, nb3, lane), acc3);
    }
    // y part: K = 512..575 of W_ih, A comes from LDS sy
    for (int k0 = 0; k0 < Nn; k0 += 32) {
        BFrag a = load_a_rowmajor(&sy[0][0] + k0, Nn, lane);
        acc0 = wmma_bf16(a, load_b_kmajor(WihT, 4 * Dd, Dd + k0, nb0, lane), acc0);
        acc1 = wmma_bf16(a, load_b_kmajor(WihT, 4 * Dd, Dd + k0, nb1, lane), acc1);
        acc2 = wmma_bf16(a, load_b_kmajor(WihT, 4 * Dd, Dd + k0, nb2, lane), acc2);
        acc3 = wmma_bf16(a, load_b_kmajor(WihT, 4 * Dd, Dd + k0, nb3, lane), acc3);
    }
    // h part: K = 0..511 of W_hh
    const __bf16* htile = hread + (size_t)rb * Dd;
    for (int k0 = 0; k0 < Dd; k0 += 32) {
        BFrag a = load_a_rowmajor(htile + k0, Dd, lane);
        acc0 = wmma_bf16(a, load_b_kmajor(WhhT, 4 * Dd, k0, nb0, lane), acc0);
        acc1 = wmma_bf16(a, load_b_kmajor(WhhT, 4 * Dd, k0, nb1, lane), acc1);
        acc2 = wmma_bf16(a, load_b_kmajor(WhhT, 4 * Dd, k0, nb2, lane), acc2);
        acc3 = wmma_bf16(a, load_b_kmajor(WhhT, 4 * Dd, k0, nb3, lane), acc3);
    }

    // Stage 3: bias + activations + cell update (C/D layout: lane L -> col = L%16, rows r+(L<16?0:8))
    int nc   = lane & 15;
    int dcol = db + wave * 16 + nc;
    float bI = bsum[0 * Dd + dcol];
    float bF = bsum[1 * Dd + dcol];
    float bG = bsum[2 * Dd + dcol];
    float bO = bsum[3 * Dd + dcol];
    int mbase = (lane & 16) ? 8 : 0;
#pragma unroll
    for (int r = 0; r < 8; ++r) {
        int row = rb + mbase + r;
        float iv = sigmoid_f(acc0[r] + bI);
        float fv = sigmoid_f(acc1[r] + bF);
        float gv = tanh_f(acc2[r] + bG);
        float ov = sigmoid_f(acc3[r] + bO);
        float cp = cread[(size_t)row * Dd + dcol];
        float cn = fv * cp + iv * gv;
        float hv = ov * tanh_f(cn);
        cwrite[(size_t)row * Dd + dcol] = cn;
        hwrite[(size_t)row * Dd + dcol] = (__bf16)hv;
    }
}

// Emit y(T-1) from the final hidden state.
__global__ void __launch_bounds__(128)
final_y_kernel(const __bf16* __restrict__ hread, const __bf16* __restrict__ WlinT,
               const float* __restrict__ b_lin, float* __restrict__ out)
{
    __shared__ float  slog[16][64];
    __shared__ __align__(16) __bf16 sy[16][64];
    int tid = threadIdx.x, lane = tid & 31, wave = tid >> 5;
    int rb = blockIdx.x * 16;
    compute_y_tile(hread, WlinT, b_lin, rb, wave, lane, tid, slog, sy);
    for (int i = tid; i < 16 * 64; i += 128) {
        int m = i >> 6, n = i & 63;
        out[((size_t)(rb + m) * Tt + (Tt - 1)) * Nn + n] = slog[m][n];
    }
}

// ---------------- preprocessing ----------------
__global__ void cvt_bf16_kernel(const float* __restrict__ src, __bf16* __restrict__ dst, int n) {
    int i = blockIdx.x * blockDim.x + threadIdx.x;
    if (i < n) dst[i] = (__bf16)src[i];
}

__global__ void transpose_bf16_kernel(const float* __restrict__ src, __bf16* __restrict__ dst,
                                      int rows, int cols) {
    int i = blockIdx.x * blockDim.x + threadIdx.x;
    if (i < rows * cols) {
        int r = i / cols, c = i % cols;
        dst[(size_t)c * rows + r] = (__bf16)src[i];
    }
}

__global__ void init_state_kernel(const float* __restrict__ init_h, const float* __restrict__ b_ih,
                                  const float* __restrict__ b_hh,
                                  __bf16* __restrict__ h0, float* __restrict__ c0,
                                  float* __restrict__ bsum) {
    int i = blockIdx.x * blockDim.x + threadIdx.x;
    if (i < Bb * Dd) { h0[i] = (__bf16)init_h[i]; c0[i] = 0.0f; }
    if (i < 4 * Dd)  { bsum[i] = b_ih[i] + b_hh[i]; }
}

extern "C" void kernel_launch(void* const* d_in, const int* in_sizes, int n_in,
                              void* d_out, int out_size, void* d_ws, size_t ws_size,
                              hipStream_t stream) {
    (void)in_sizes; (void)n_in; (void)out_size; (void)ws_size;
    const float* x      = (const float*)d_in[0];
    const float* init_h = (const float*)d_in[1];
    const float* W_ih   = (const float*)d_in[2];
    const float* b_ih   = (const float*)d_in[3];
    const float* W_hh   = (const float*)d_in[4];
    const float* b_hh   = (const float*)d_in[5];
    const float* W_lin  = (const float*)d_in[6];
    const float* b_lin  = (const float*)d_in[7];
    float* out = (float*)d_out;

    char* ws = (char*)d_ws;
    size_t off = 0;
    auto alloc = [&](size_t bytes) -> void* {
        void* p = ws + off;
        off = (off + bytes + 255) & ~(size_t)255;
        return p;
    };
    __bf16* xbf   = (__bf16*)alloc((size_t)Bb * Tt * Dd * 2);     // 128 MB
    __bf16* WihT  = (__bf16*)alloc((size_t)(Dd + Nn) * 4 * Dd * 2);
    __bf16* WhhT  = (__bf16*)alloc((size_t)Dd * 4 * Dd * 2);
    __bf16* WlinT = (__bf16*)alloc((size_t)Dd * Nn * 2);
    float*  bsum  = (float*)alloc((size_t)4 * Dd * 4);
    __bf16* hbuf0 = (__bf16*)alloc((size_t)Bb * Dd * 2);
    __bf16* hbuf1 = (__bf16*)alloc((size_t)Bb * Dd * 2);
    float*  cbuf0 = (float*)alloc((size_t)Bb * Dd * 4);
    float*  cbuf1 = (float*)alloc((size_t)Bb * Dd * 4);

    int nx = Bb * Tt * Dd;
    cvt_bf16_kernel<<<(nx + 255) / 256, 256, 0, stream>>>(x, xbf, nx);
    transpose_bf16_kernel<<<(4 * Dd * (Dd + Nn) + 255) / 256, 256, 0, stream>>>(W_ih, WihT, 4 * Dd, Dd + Nn);
    transpose_bf16_kernel<<<(4 * Dd * Dd + 255) / 256, 256, 0, stream>>>(W_hh, WhhT, 4 * Dd, Dd);
    transpose_bf16_kernel<<<(Nn * Dd + 255) / 256, 256, 0, stream>>>(W_lin, WlinT, Nn, Dd);
    init_state_kernel<<<(Bb * Dd + 255) / 256, 256, 0, stream>>>(init_h, b_ih, b_hh, hbuf0, cbuf0, bsum);

    __bf16* hb[2] = {hbuf0, hbuf1};
    float*  cb[2] = {cbuf0, cbuf1};
    for (int t = 0; t < Tt; ++t) {
        int rd = t & 1, wr = rd ^ 1;
        lstm_step_kernel<<<dim3(Bb / 16, Dd / 64), 128, 0, stream>>>(
            xbf, WihT, WhhT, WlinT, bsum, b_lin,
            hb[rd], hb[wr], cb[rd], cb[wr], out, t);
    }
    final_y_kernel<<<Bb / 16, 128, 0, stream>>>(hb[Tt & 1], WlinT, b_lin, out);
}
